// SSM_49306224558206
// MI455X (gfx1250) — compile-verified, matrix operand order checked by reading, not run on hardware
//
#include <hip/hip_runtime.h>
#include <hip/hip_bf16.h>

typedef __bf16 bf16_t;
typedef __attribute__((ext_vector_type(8)))  __bf16 v8bf;
typedef __attribute__((ext_vector_type(16))) __bf16 v16bf;
typedef __attribute__((ext_vector_type(8)))  float  v8f;
typedef __attribute__((ext_vector_type(4)))  int    v4i;

#define MAX_EFF 0.99f
#define EPS_N   1e-8f

#define BATCH   64
#define CPD     32
#define FRAMES  256
#define IN_DIM  1024
#define ST_DIM  512
#define MROWS   (BATCH * FRAMES)   // 16384
#define NSAMP   131072

union U16 { v16bf v; v8bf h[2]; };

// ---------------------------------------------------------------------------
// CDNA5 async global->LDS copy (guarded; falls back to load+store).
// ---------------------------------------------------------------------------
#if defined(__has_builtin)
#if __has_builtin(__builtin_amdgcn_global_load_async_to_lds_b128)
#define HAVE_ASYNC_B128 1
#endif
#if __has_builtin(__builtin_amdgcn_s_wait_asynccnt)
#define HAVE_WAIT_ASYNC 1
#endif
#endif

typedef __attribute__((address_space(1))) v4i* gas1_v4i;
typedef __attribute__((address_space(3))) v4i* gas3_v4i;

__device__ __forceinline__ void async_copy16(const bf16_t* g, bf16_t* l) {
#ifdef HAVE_ASYNC_B128
  __builtin_amdgcn_global_load_async_to_lds_b128(
      (gas1_v4i)(unsigned long long)(uintptr_t)g,
      (gas3_v4i)(unsigned)(uintptr_t)l, 0, 0);
#else
  *(v8bf*)l = *(const v8bf*)g;
#endif
}

__device__ __forceinline__ void async_wait() {
#ifdef HAVE_ASYNC_B128
#ifdef HAVE_WAIT_ASYNC
  __builtin_amdgcn_s_wait_asynccnt(0);
#else
  asm volatile("s_wait_asynccnt 0" ::: "memory");
#endif
#endif
}

// ---------------------------------------------------------------------------
// Weight f32 -> bf16 transposed convert:  Wt[n*K + k] = W[k*N + n]
// ---------------------------------------------------------------------------
__global__ void k_convT(const float* __restrict__ W, bf16_t* __restrict__ Wt,
                        int K, int N) {
  int idx = blockIdx.x * blockDim.x + threadIdx.x;
  if (idx >= K * N) return;
  int k = idx / N, n = idx % N;
  Wt[(size_t)n * K + k] = (bf16_t)W[idx];
}

// ---------------------------------------------------------------------------
// p = limit(ctrl' @ proj) ; K=32 -> scalar FMA, one block per row.
// ---------------------------------------------------------------------------
__global__ __launch_bounds__(128) void k_proj(const float* __restrict__ control,
                       const float* __restrict__ proj,
                       bf16_t* __restrict__ p, float* __restrict__ pnorm) {
  __shared__ float sv[CPD];
  __shared__ float red[128];
  const int row = blockIdx.x;
  const int bat = row / FRAMES, f = row % FRAMES;
  const int tid = threadIdx.x;
  if (tid < CPD)
    sv[tid] = control[(size_t)bat * CPD * FRAMES + (size_t)tid * FRAMES + f];
  __syncthreads();
  float on2 = 0.f;
  if (tid < CPD) { float v = sv[tid]; on2 = v * v; }
  red[tid] = on2; __syncthreads();
  for (int s = 64; s > 0; s >>= 1) { if (tid < s) red[tid] += red[tid + s]; __syncthreads(); }
  const float orig = sqrtf(red[0]);
  __syncthreads();

  float x[8], ss = 0.f;
#pragma unroll
  for (int i = 0; i < 8; ++i) {
    const int n = tid * 8 + i;
    float acc = 0.f;
#pragma unroll
    for (int c = 0; c < CPD; ++c) acc += sv[c] * proj[(size_t)c * IN_DIM + n];
    x[i] = acc; ss += acc * acc;
  }
  red[tid] = ss; __syncthreads();
  for (int s = 64; s > 0; s >>= 1) { if (tid < s) red[tid] += red[tid + s]; __syncthreads(); }
  const float nn = sqrtf(red[0]);
  const float clamped = fminf(nn, orig * MAX_EFF);
  const float scale = clamped / (nn + EPS_N);
#pragma unroll
  for (int i = 0; i < 8; ++i)
    p[(size_t)row * IN_DIM + tid * 8 + i] = (bf16_t)(x[i] * scale);
  if (tid == 0) pnorm[row] = clamped;
}

// ---------------------------------------------------------------------------
// bf16 WMMA GEMM: X[M,N] = A[M,K] @ Bt[N,K].  8 waves/block; block tile
// 32(M) x 512(N); each wave: 2 M-tiles x 4 N-tiles = 8 wmma per k-step
// against 12 loads. A tile staged in LDS via async-to-LDS, rows padded.
// ---------------------------------------------------------------------------
template <int K>
__global__ __launch_bounds__(256) void k_gemm_bf16(const bf16_t* __restrict__ A,
                                                   const bf16_t* __restrict__ Bt,
                                                   float* __restrict__ X, int N) {
  constexpr int LDA = K + 8;                 // pad: spread LDS banks
  __shared__ bf16_t sA[32 * LDA];
  const int tid   = threadIdx.x;
  const int lane  = tid & 31;
  const int wave  = tid >> 5;
  const int mbase = blockIdx.y * 32;
  const int lh    = lane & 15;
  const int lo    = (lane >= 16) ? 8 : 0;
  const int rb    = (lane >= 16) ? 8 : 0;

  {  // cooperative async stage of A[mbase .. +31][0..K) into LDS, 16B chunks
    constexpr int CPR = K / 8;
    const bf16_t* ag = A + (size_t)mbase * K;
    for (int c = tid; c < 32 * CPR; c += 256) {
      const int r = c / CPR, kc = (c % CPR) * 8;
      async_copy16(ag + (size_t)r * K + kc, &sA[r * LDA + kc]);
    }
    async_wait();
  }
  __syncthreads();

  const int ntbase = (blockIdx.x * 8 + wave) * 64;
  const bf16_t* br0 = Bt + (size_t)(ntbase +  0 + lh) * K;
  const bf16_t* br1 = Bt + (size_t)(ntbase + 16 + lh) * K;
  const bf16_t* br2 = Bt + (size_t)(ntbase + 32 + lh) * K;
  const bf16_t* br3 = Bt + (size_t)(ntbase + 48 + lh) * K;
  const bf16_t* arow0 = &sA[lh * LDA];
  const bf16_t* arow1 = &sA[(16 + lh) * LDA];

  v8f c00 = {}, c01 = {}, c02 = {}, c03 = {};
  v8f c10 = {}, c11 = {}, c12 = {}, c13 = {};
  for (int kb = 0; kb < K; kb += 32) {
    U16 a0, a1, b0, b1, b2, b3;
    a0.h[0] = *(const v8bf*)(arow0 + kb + lo);
    a0.h[1] = *(const v8bf*)(arow0 + kb + lo + 16);
    a1.h[0] = *(const v8bf*)(arow1 + kb + lo);
    a1.h[1] = *(const v8bf*)(arow1 + kb + lo + 16);
    b0.h[0] = *(const v8bf*)(br0 + kb + lo);
    b0.h[1] = *(const v8bf*)(br0 + kb + lo + 16);
    b1.h[0] = *(const v8bf*)(br1 + kb + lo);
    b1.h[1] = *(const v8bf*)(br1 + kb + lo + 16);
    b2.h[0] = *(const v8bf*)(br2 + kb + lo);
    b2.h[1] = *(const v8bf*)(br2 + kb + lo + 16);
    b3.h[0] = *(const v8bf*)(br3 + kb + lo);
    b3.h[1] = *(const v8bf*)(br3 + kb + lo + 16);
    __builtin_prefetch(br0 + kb + 512, 0, 1);
    __builtin_prefetch(br2 + kb + 512, 0, 1);
    c00 = __builtin_amdgcn_wmma_f32_16x16x32_bf16(false, a0.v, false, b0.v, (short)0, c00, false, false);
    c01 = __builtin_amdgcn_wmma_f32_16x16x32_bf16(false, a0.v, false, b1.v, (short)0, c01, false, false);
    c02 = __builtin_amdgcn_wmma_f32_16x16x32_bf16(false, a0.v, false, b2.v, (short)0, c02, false, false);
    c03 = __builtin_amdgcn_wmma_f32_16x16x32_bf16(false, a0.v, false, b3.v, (short)0, c03, false, false);
    c10 = __builtin_amdgcn_wmma_f32_16x16x32_bf16(false, a1.v, false, b0.v, (short)0, c10, false, false);
    c11 = __builtin_amdgcn_wmma_f32_16x16x32_bf16(false, a1.v, false, b1.v, (short)0, c11, false, false);
    c12 = __builtin_amdgcn_wmma_f32_16x16x32_bf16(false, a1.v, false, b2.v, (short)0, c12, false, false);
    c13 = __builtin_amdgcn_wmma_f32_16x16x32_bf16(false, a1.v, false, b3.v, (short)0, c13, false, false);
  }

  float* x0 = X + (size_t)(mbase + rb) * N + lh;
  float* x1 = X + (size_t)(mbase + 16 + rb) * N + lh;
#pragma unroll
  for (int i = 0; i < 8; ++i) {
    x0[(size_t)i * N + ntbase +  0] = c00[i];
    x0[(size_t)i * N + ntbase + 16] = c01[i];
    x0[(size_t)i * N + ntbase + 32] = c02[i];
    x0[(size_t)i * N + ntbase + 48] = c03[i];
    x1[(size_t)i * N + ntbase +  0] = c10[i];
    x1[(size_t)i * N + ntbase + 16] = c11[i];
    x1[(size_t)i * N + ntbase + 32] = c12[i];
    x1[(size_t)i * N + ntbase + 48] = c13[i];
  }
}

// ---------------------------------------------------------------------------
// In-place norm limit.
// ---------------------------------------------------------------------------
__global__ __launch_bounds__(256) void k_limit(float* __restrict__ X,
                        const float* __restrict__ vnorm, int N) {
  const int row = blockIdx.x, tid = threadIdx.x;
  float* xr = X + (size_t)row * N;
  float ss = 0.f;
  for (int n = tid; n < N; n += 256) { float v = xr[n]; ss += v * v; }
  __shared__ float red[256];
  red[tid] = ss; __syncthreads();
  for (int s = 128; s > 0; s >>= 1) { if (tid < s) red[tid] += red[tid + s]; __syncthreads(); }
  const float nn = sqrtf(red[0]);
  const float scale = fminf(nn, vnorm[row] * MAX_EFF) / (nn + EPS_N);
  for (int n = tid; n < N; n += 256) xr[n] *= scale;
}

// ---------------------------------------------------------------------------
// Recurrence scan. Batch rows are INDEPENDENT: 4 blocks x 16 rows each.
// 1024 threads = 32 waves = 32 N-tiles of the (16 x 512) state update.
// B operand: first 8 k-steps resident in registers (64 VGPRs); last 8
// k-steps re-read from global each frame via an opaque ZERO OFFSET (not an
// opaque pointer: that stripped the inferred address space and produced
// flat_load, which ties up LDS issue alongside our real ds_loads). The
// integer launder keeps global_load while still defeating LICM.
// Row-norm reductions: non-atomic LDS scatter + 2-stage fold.
// ---------------------------------------------------------------------------
#define SC_LDA (ST_DIM + 8)
#define SQ_C   (16 * 16 + 4)                 // per-wave sq stride (pad banks)
__global__ __launch_bounds__(1024, 1) void k_scan(const bf16_t* __restrict__ smT,
                       const float* __restrict__ b_f32,
                       bf16_t* __restrict__ sprime,
                       float* __restrict__ snorm) {
  __shared__ bf16_t sstate[16 * SC_LDA];     // bf16 state tile, padded rows
  __shared__ float  sq[32 * SQ_C];           // [wave][row][lane-half] squares
  __shared__ float  warr[16 * 32];           // [row][wave] partials
  __shared__ float  sscale[16];
  __shared__ float  son2[2][16];             // ||state||^2 double buffer

  const int tid   = threadIdx.x;
  const int lane  = tid & 31;
  const int wv    = tid >> 5;                // 0..31 = N tile
  const int lh    = lane & 15;
  const int lo    = (lane >= 16) ? 8 : 0;
  const int rb    = (lane >= 16) ? 8 : 0;
  const int grp   = blockIdx.x;              // batch rows grp*16 .. +15
  const int bcol  = wv * 16 + lh;
  const bf16_t* brow = smT + (size_t)bcol * ST_DIM;
  const bf16_t* arow = &sstate[lh * SC_LDA];
  const int rsq = tid >> 5, csq = tid & 31;  // stage-A fold mapping (tid<512)

  // B operand: half resident in registers for all 256 steps
  U16 breg[8];
#pragma unroll
  for (int kk = 0; kk < 8; ++kk) {
    breg[kk].h[0] = *(const v8bf*)(brow + kk * 32 + lo);
    breg[kk].h[1] = *(const v8bf*)(brow + kk * 32 + lo + 16);
  }

  for (int i = tid; i < 16 * SC_LDA; i += 1024) sstate[i] = (bf16_t)0.f;
  if (tid < 16) { son2[0][tid] = 0.f; son2[1][tid] = 0.f; }
  __syncthreads();

  int opq = 0;                               // laundered zero offset
  for (int t = 0; t < FRAMES; ++t) {
    // Opaque per-iteration offset: blocks LICM from hoisting the 8
    // second-half B loads (which would force spills) while keeping the
    // base pointer's global address space (-> global_load, not flat).
    asm volatile("" : "+v"(opq));
    const bf16_t* bp = brow + 8 * 32 + opq;
    // ---- s'_pre = s @ SM ----
    v8f acc = {};
#pragma unroll
    for (int kk = 0; kk < 8; ++kk) {         // register-resident B
      U16 a;
      a.h[0] = *(const v8bf*)(arow + kk * 32 + lo);
      a.h[1] = *(const v8bf*)(arow + kk * 32 + lo + 16);
      acc = __builtin_amdgcn_wmma_f32_16x16x32_bf16(false, a.v, false, breg[kk].v,
                                                    (short)0, acc, false, false);
    }
#pragma unroll
    for (int kk = 0; kk < 8; ++kk) {         // B from global (L0-resident)
      U16 a, b;
      a.h[0] = *(const v8bf*)(arow + (8 + kk) * 32 + lo);
      a.h[1] = *(const v8bf*)(arow + (8 + kk) * 32 + lo + 16);
      b.h[0] = *(const v8bf*)(bp + kk * 32 + lo);
      b.h[1] = *(const v8bf*)(bp + kk * 32 + lo + 16);
      acc = __builtin_amdgcn_wmma_f32_16x16x32_bf16(false, a.v, false, b.v,
                                                    (short)0, acc, false, false);
    }
    // ---- scatter squares (private slots, no conflicts) ----
#pragma unroll
    for (int i = 0; i < 8; ++i)
      sq[wv * SQ_C + (rb + i) * 16 + lh] = acc[i] * acc[i];
    __syncthreads();
    if (tid < 512) {                           // fold over 16 lane-slots
      const float4* q = (const float4*)&sq[csq * SQ_C + rsq * 16];
      float4 q0 = q[0], q1 = q[1], q2 = q[2], q3 = q[3];
      warr[rsq * 32 + csq] = ((q0.x + q0.y) + (q0.z + q0.w)) + ((q1.x + q1.y) + (q1.z + q1.w))
                           + ((q2.x + q2.y) + (q2.z + q2.w)) + ((q3.x + q3.y) + (q3.z + q3.w));
    }
    __syncthreads();
    if (tid < 16) {                            // fold over 32 waves; norms
      float nn2 = 0.f;
#pragma unroll
      for (int w = 0; w < 32; ++w) nn2 += warr[tid * 32 + w];
      const float nn = sqrtf(nn2);
      const float on = sqrtf(son2[t & 1][tid]);
      const float cl = fminf(nn, on * MAX_EFF);
      sscale[tid] = cl / (nn + EPS_N);
      snorm[t * BATCH + grp * 16 + tid] = cl;
    }
    __syncthreads();
    // ---- scale, emit s', update state with +b_t ----
#pragma unroll
    for (int i = 0; i < 8; ++i) {
      const int r = rb + i;
      const int gr = grp * 16 + r;
      const float sv = acc[i] * sscale[r];
      sprime[((size_t)t * BATCH + gr) * ST_DIM + bcol] = (bf16_t)sv;
      const float ns = sv + b_f32[((size_t)gr * FRAMES + t) * ST_DIM + bcol];
      sstate[r * SC_LDA + bcol] = (bf16_t)ns;
      sq[wv * SQ_C + r * 16 + lh] = ns * ns;
    }
    __syncthreads();
    if (tid < 512) {
      const float4* q = (const float4*)&sq[csq * SQ_C + rsq * 16];
      float4 q0 = q[0], q1 = q[1], q2 = q[2], q3 = q[3];
      warr[rsq * 32 + csq] = ((q0.x + q0.y) + (q0.z + q0.w)) + ((q1.x + q1.y) + (q1.z + q1.w))
                           + ((q2.x + q2.y) + (q2.z + q2.w)) + ((q3.x + q3.y) + (q3.z + q3.w));
    }
    __syncthreads();
    if (tid < 16) {                            // ||next state||^2
      float o2 = 0.f;
#pragma unroll
      for (int w = 0; w < 32; ++w) o2 += warr[tid * 32 + w];
      son2[(t + 1) & 1][tid] = o2;
    }
    // no barrier needed here: son2 written+read by the same wave-0 threads,
    // and warr/sq are only rewritten after the next iteration's barrier.
  }
}

// ---------------------------------------------------------------------------
// out[b][x] = w[j]*(c_f[j]+d_f[j]) + w[j+512]*(c_{f-1}[j+512]+d_{f-1}[j+512])
// ---------------------------------------------------------------------------
__global__ __launch_bounds__(256) void k_final(const float* __restrict__ c,
                        const float* __restrict__ d, float* __restrict__ out) {
  const size_t idx = (size_t)blockIdx.x * blockDim.x + threadIdx.x;
  const int b = (int)(idx >> 17);
  const int x = (int)(idx & (NSAMP - 1));
  const int f = x >> 9;
  const int j = x & 511;
  const float k = 6.28318530717958647f / 1023.0f;
  const float w1 = 0.54f - 0.46f * __cosf(k * (float)j);
  float val = w1 * (c[((size_t)f * BATCH + b) * IN_DIM + j] +
                    d[((size_t)b * FRAMES + f) * IN_DIM + j]);
  if (f >= 1) {
    const int j2 = j + 512;
    const float w2 = 0.54f - 0.46f * __cosf(k * (float)j2);
    val += w2 * (c[((size_t)(f - 1) * BATCH + b) * IN_DIM + j2] +
                 d[((size_t)b * FRAMES + (f - 1)) * IN_DIM + j2]);
  }
  out[idx] = val;
}

// ---------------------------------------------------------------------------
extern "C" void kernel_launch(void* const* d_in, const int* in_sizes, int n_in,
                              void* d_out, int out_size, void* d_ws, size_t ws_size,
                              hipStream_t stream) {
  const float* control = (const float*)d_in[0];
  const float* proj    = (const float*)d_in[1];
  const float* sm      = (const float*)d_in[2];
  const float* im      = (const float*)d_in[3];
  const float* om      = (const float*)d_in[4];
  const float* dm      = (const float*)d_in[5];
  float* out = (float*)d_out;

  char* ws = (char*)d_ws;
  size_t off = 0;
  auto alloc = [&](size_t bytes) -> char* {
    char* p = ws + off;
    off = (off + bytes + 255) & ~(size_t)255;
    return p;
  };

  bf16_t* inT    = (bf16_t*)alloc((size_t)ST_DIM * IN_DIM * 2);
  bf16_t* smT    = (bf16_t*)alloc((size_t)ST_DIM * ST_DIM * 2);
  bf16_t* omT    = (bf16_t*)alloc((size_t)IN_DIM * ST_DIM * 2);
  bf16_t* dmT    = (bf16_t*)alloc((size_t)IN_DIM * IN_DIM * 2);
  bf16_t* p_bf   = (bf16_t*)alloc((size_t)MROWS * IN_DIM * 2);
  float*  pnorm  = (float*) alloc((size_t)MROWS * 4);
  float*  Xb     = (float*) alloc((size_t)MROWS * ST_DIM * 4);
  float*  Xd     = (float*) alloc((size_t)MROWS * IN_DIM * 4);
  bf16_t* sprime = (bf16_t*)alloc((size_t)MROWS * ST_DIM * 2);
  float*  snorm  = (float*) alloc((size_t)MROWS * 4);
  float*  Xc     = (float*) alloc((size_t)MROWS * IN_DIM * 4);

  // weight converts (f32 -> transposed bf16)
  k_convT<<<(IN_DIM * ST_DIM + 255) / 256, 256, 0, stream>>>(im, inT, IN_DIM, ST_DIM);
  k_convT<<<(ST_DIM * ST_DIM + 255) / 256, 256, 0, stream>>>(sm, smT, ST_DIM, ST_DIM);
  k_convT<<<(ST_DIM * IN_DIM + 255) / 256, 256, 0, stream>>>(om, omT, ST_DIM, IN_DIM);
  k_convT<<<(IN_DIM * IN_DIM + 255) / 256, 256, 0, stream>>>(dm, dmT, IN_DIM, IN_DIM);

  // p = limit(ctrl @ proj)
  k_proj<<<MROWS, 128, 0, stream>>>(control, proj, p_bf, pnorm);

  // b = limit(p @ input_matrix)
  k_gemm_bf16<IN_DIM><<<dim3(ST_DIM / 512, MROWS / 32), 256, 0, stream>>>(p_bf, inT, Xb, ST_DIM);
  k_limit<<<MROWS, 256, 0, stream>>>(Xb, pnorm, ST_DIM);

  // d = limit(p @ direct_matrix)
  k_gemm_bf16<IN_DIM><<<dim3(IN_DIM / 512, MROWS / 32), 256, 0, stream>>>(p_bf, dmT, Xd, IN_DIM);
  k_limit<<<MROWS, 256, 0, stream>>>(Xd, pnorm, IN_DIM);

  // sequential recurrence: 4 independent blocks (16 batch rows each)
  k_scan<<<4, 1024, 0, stream>>>(smT, Xb, sprime, snorm);

  // c = limit(s' @ output_matrix)
  k_gemm_bf16<ST_DIM><<<dim3(IN_DIM / 512, MROWS / 32), 256, 0, stream>>>(sprime, omT, Xc, IN_DIM);
  k_limit<<<MROWS, 256, 0, stream>>>(Xc, snorm, IN_DIM);

  // epilogue: c + d, window, overlap-add
  k_final<<<(size_t)BATCH * NSAMP / 256, 256, 0, stream>>>(Xc, Xd, out);
}
